// FP8LinearWrapper_20693152432189
// MI455X (gfx1250) — matrix operand order repaired and checked by reading, not compiled, not run on hardware
//
#include <hip/hip_runtime.h>

// ---------------------------------------------------------------------------
// FP8 (weight-only) linear:  out[m,n] = sum_k x[m,k] * (fp8->f32(w[n,k]) * s)
//                                     + bias[n]
// M = 8192, N = 4096, K = 4096.
// bf16 WMMA (v_wmma_f32_16x16x32_bf16), f32 accumulate.
//   * fp8 e4m3 -> bf16 exact; inv_scale folded into f32 epilogue (exact).
//   * only rounding vs reference: x f32 -> bf16.
// Block: 256 threads (8 waves, 2M x 4N), block tile 128x256, wave tile 64x64.
// Double-buffered LDS + register prefetch of the next K-chunk.
// ---------------------------------------------------------------------------

#define BM 128
#define BN 256
#define BK 64
#define LDK 72   // padded LDS row (bf16 elems): 144B stride -> conflict-free b128

typedef __attribute__((ext_vector_type(16))) __bf16 v16bf;
typedef __attribute__((ext_vector_type(8)))  __bf16 v8bf;
typedef __attribute__((ext_vector_type(8)))  float  v8f;

// Exact e4m3fn -> f32: sign to bit31, exp+mant to bits [26:20]; exponent then
// sits in an 8-bit field biased 127 instead of 7 -> fix with one mul by 2^120
// (exact for all normals and denormals).
__device__ __forceinline__ float fp8_e4m3_to_f32(unsigned int b) {
    union { unsigned int u; float f; } v;
    v.u = ((b & 0x80u) << 24) | ((b & 0x7fu) << 20);
    return v.f * 0x1.0p+120f;
}

__global__ __launch_bounds__(256)
void fp8_linear_wmma_bf16(const float* __restrict__ X,           // [M,K] f32
                          const unsigned char* __restrict__ W,   // [N,K] fp8
                          const float* __restrict__ inv_scale_p, // [1]
                          const float* __restrict__ bias,        // [N]
                          float* __restrict__ Out,               // [M,N]
                          int M, int N, int K) {
    __shared__ __bf16 As[2][BM][LDK];
    __shared__ __bf16 Bs[2][BN][LDK];

    const int tid  = threadIdx.x;
    const int lane = tid & 31;
    const int wave = tid >> 5;
    const int r    = lane & 15;   // lane within half-wave
    const int half = lane >> 4;   // 0 or 1

    const int waveM = wave >> 2;  // 0..1 -> 64 rows each
    const int waveN = wave & 3;   // 0..3 -> 64 cols each

    const int blockM = blockIdx.y * BM;
    const int blockN = blockIdx.x * BN;

    const float inv_scale = inv_scale_p[0];

    // ---- per-thread staging geometry (constant across chunks) -------------
    const int arow = tid >> 4;          // + it*16   (16 rows / iter, 8 iters)
    const int acol = (tid & 15) * 4;    // 4 floats per thread per row
    const int brow = tid >> 2;          // + it*64   (64 rows / iter, 4 iters)
    const int bcol = (tid & 3) * 16;    // 16 fp8 bytes per thread per row

    const float*         aptr = X + (size_t)(blockM + arow) * K + acol;
    const unsigned char* bptr = W + (size_t)(blockN + brow) * K + bcol;

    v8f acc[4][4];
#pragma unroll
    for (int i = 0; i < 4; ++i)
#pragma unroll
        for (int j = 0; j < 4; ++j)
#pragma unroll
            for (int g = 0; g < 8; ++g) acc[i][j][g] = 0.0f;

    float4 a_reg[8];
    uint4  b_reg[4];

    // -------- prefetch: global -> registers for chunk at K offset kk ----
    auto prefetch = [&](int kk) {
#pragma unroll
        for (int it = 0; it < 8; ++it)
            a_reg[it] = *(const float4*)(aptr + (size_t)(it * 16) * K + kk);
#pragma unroll
        for (int it = 0; it < 4; ++it)
            b_reg[it] = *(const uint4*)(bptr + (size_t)(it * 64) * K + kk);
    };

    // -------- convert registers -> LDS buffer `buf` ------------------------
    auto cvt_store = [&](int buf) {
#pragma unroll
        for (int it = 0; it < 8; ++it) {
            __bf16* dst = &As[buf][it * 16 + arow][acol];
            dst[0] = (__bf16)a_reg[it].x;
            dst[1] = (__bf16)a_reg[it].y;
            dst[2] = (__bf16)a_reg[it].z;
            dst[3] = (__bf16)a_reg[it].w;
        }
#pragma unroll
        for (int it = 0; it < 4; ++it) {
            unsigned int wds[4] = {b_reg[it].x, b_reg[it].y,
                                   b_reg[it].z, b_reg[it].w};
            __bf16* dst = &Bs[buf][it * 64 + brow][bcol];
#pragma unroll
            for (int wd = 0; wd < 4; ++wd)
#pragma unroll
                for (int by = 0; by < 4; ++by)
                    dst[wd * 4 + by] =
                        (__bf16)fp8_e4m3_to_f32((wds[wd] >> (8 * by)) & 0xffu);
        }
    };

    // -------- compute one K-chunk out of LDS buffer `buf` ------------------
    auto compute = [&](int buf) {
#pragma unroll
        for (int ks = 0; ks < 2; ++ks) {
            // B fragments (32x16): lanes 0-15 -> K 0-15, lanes 16-31 -> K 16-31
            v16bf bfrag[4];
#pragma unroll
            for (int j = 0; j < 4; ++j) {
                const int ncol = waveN * 64 + j * 16 + r;
                const __bf16* p = &Bs[buf][ncol][ks * 32 + half * 16];
                v8bf lo = *(const v8bf*)p;
                v8bf hi = *(const v8bf*)(p + 8);
                bfrag[j] = __builtin_shufflevector(
                    lo, hi, 0,1,2,3,4,5,6,7,8,9,10,11,12,13,14,15);
            }
#pragma unroll
            for (int i = 0; i < 4; ++i) {
                // A fragment (16x32): lanes 0-15 -> K 0-15,
                //                     lanes 16-31 -> K 16-31
                const int mrow = waveM * 64 + i * 16 + r;
                const __bf16* p = &As[buf][mrow][ks * 32 + half * 16];
                v8bf lo = *(const v8bf*)p;
                v8bf hi = *(const v8bf*)(p + 8);
                v16bf afrag = __builtin_shufflevector(
                    lo, hi, 0,1,2,3,4,5,6,7,8,9,10,11,12,13,14,15);
#pragma unroll
                for (int j = 0; j < 4; ++j) {
                    acc[i][j] = __builtin_amdgcn_wmma_f32_16x16x32_bf16(
                        false, afrag, false, bfrag[j],
                        (short)0, acc[i][j], false, false);
                }
            }
        }
    };

    // -------- software pipeline -------------------------------------------
    const int NCHUNK = K / BK;
    prefetch(0);
    cvt_store(0);
    __syncthreads();

    for (int c = 0; c < NCHUNK; ++c) {
        const int cur = c & 1;
        if (c + 1 < NCHUNK) prefetch((c + 1) * BK);  // HBM in flight over WMMAs
        compute(cur);
        if (c + 1 < NCHUNK) cvt_store(cur ^ 1);      // fill other buffer
        __syncthreads();
    }

    // -------- epilogue: D = acc * inv_scale + bias -------------------------
    // C/D layout: VGPR g -> row (g + half*8), lanes 0-15 -> col r.
#pragma unroll
    for (int i = 0; i < 4; ++i) {
        const int mbase = blockM + waveM * 64 + i * 16 + half * 8;
#pragma unroll
        for (int j = 0; j < 4; ++j) {
            const int nb = blockN + waveN * 64 + j * 16 + r;
            const float bval = bias[nb];
#pragma unroll
            for (int g = 0; g < 8; ++g) {
                Out[(size_t)(mbase + g) * N + nb] =
                    acc[i][j][g] * inv_scale + bval;
            }
        }
    }
}

extern "C" void kernel_launch(void* const* d_in, const int* in_sizes, int n_in,
                              void* d_out, int out_size, void* d_ws, size_t ws_size,
                              hipStream_t stream) {
    const float*         X         = (const float*)d_in[0];
    const unsigned char* W         = (const unsigned char*)d_in[1];  // fp8 e4m3
    const float*         inv_scale = (const float*)d_in[2];
    const float*         bias      = (const float*)d_in[3];
    float*               Out       = (float*)d_out;

    const int M = 4 * 2048, N = 4096, K = 4096;
    dim3 grid(N / BN, M / BM);   // (16, 64)
    fp8_linear_wmma_bf16<<<grid, dim3(256), 0, stream>>>(
        X, W, inv_scale, bias, Out, M, N, K);
}
